// Channel_CAM_38826504356088
// MI455X (gfx1250) — matrix operand order repaired and checked by reading, not compiled, not run on hardware
//
#include <hip/hip_runtime.h>
#include <hip/hip_bf16.h>
#include <cstdint>

// ---------------------------------------------------------------------------
// MI455X (gfx1250, wave32) implementation.
// Problem: B=2, H=W=256, C=256  ->  NPIX = 131072 pixels, NHWC input.
// out = Wf1 @ xn + M2 @ y   with  M2 = Wf2 @ w_up @ G0   (rank-16 fold)
// Dominant GEMM: v_wmma_f32_16x16x32_f16 with weights LDS-resident (CDNA5
// 320KB LDS), staged via the Tensor Data Mover (tensor_load_to_lds).
// ---------------------------------------------------------------------------

typedef __attribute__((ext_vector_type(16))) _Float16 v16h;
typedef __attribute__((ext_vector_type(8)))  float    v8f;
typedef __attribute__((ext_vector_type(4)))  uint32_t u32x4;
typedef __attribute__((ext_vector_type(8)))  uint32_t u32x8;

#define NPIX   131072      // B*H*W
#define HW     65536       // H*W (per sample)
#define C      256
#define BN_EPS 1e-5f

// Workspace layout (bytes). wf1f and m2f are contiguous -> one TDM descriptor.
#define WS_STATS   0          // 512 f32: [sum|mean][sumsq|inv_std]
#define WS_G       2048       // 256 f32: Gram of sample 0
#define WS_WF1F    4096       // 65536 f16 (128KB): Wf1 B-fragments (8 kt x 16 nt)
#define WS_M2F     135168     // 8192  f16 (16KB):  M2  B-fragments (1 kt x 16 nt)
#define WS_Y       262144     // NPIX*16 f32: y = w_down @ a

// A-fragment (16x32 f16) lane/element placement per ISA 7.12.2:
// lanes 0-15 hold rows M=0..15 with K {0..7,16..23}; lanes 16-31 same rows,
// K {8..15,24..31}; element h -> (pair=h/2, off=h&1), k = base(pair)+8*hi+off.
__device__ __forceinline__ void afrag_pos(int k, int m, int& kk, int& lane, int& h) {
    kk = k >> 5;
    int kr   = k & 31;
    int off  = kr & 1;
    int hi   = (kr >> 3) & 1;
    int base = kr - 8 * hi - off;              // {0,2,4,6,16,18,20,22}
    int pair = (base < 8) ? (base >> 1) : (4 + ((base - 16) >> 1));
    h    = pair * 2 + off;
    lane = m + 16 * hi;
}

__device__ __forceinline__ v8f wmma_f16(v16h a, v16h b, v8f c) {
    return __builtin_amdgcn_wmma_f32_16x16x32_f16(false, a, false, b, (short)0, c,
                                                  false, false);
}

// ---------------------------------------------------------------------------
// K0: zero the ws accumulators (stats + Gram) -- deterministic per launch.
__global__ void init_k(float* ws) {
    for (int k = threadIdx.x; k < 768; k += 256) ws[k] = 0.f;
}

// K1: per-channel sum / sum-of-squares. NHWC: channel is innermost -> coalesced.
__global__ void bn_stats_k(const float* __restrict__ x, float* __restrict__ stats) {
    const int c = threadIdx.x;
    float s = 0.f, sq = 0.f;
    for (int p = blockIdx.x; p < NPIX; p += gridDim.x) {
        float v = x[(size_t)p * C + c];
        s += v; sq += v * v;
    }
    atomicAdd(&stats[c], s);
    atomicAdd(&stats[C + c], sq);
}

// K2: finalize mean / inv_std (biased variance, training-mode BN).
__global__ void bn_finalize_k(float* stats) {
    const int c = threadIdx.x;
    const float invN = 1.f / (float)NPIX;
    float mean = stats[c] * invN;
    float var  = stats[C + c] * invN - mean * mean;
    stats[c]     = mean;
    stats[C + c] = rsqrtf(var + BN_EPS);
}

// K3: y[p,0:16] = sigmoid(relu(xn[p,:])) @ w_down^T via WMMA.
// 256 threads = 8 waves; each chunk = 128 px staged as packed A-fragments in LDS.
__global__ void down_gemm_k(const float* __restrict__ x,
                            const float* __restrict__ wdn,
                            const float* __restrict__ stats,
                            float* __restrict__ y) {
    __shared__ __attribute__((aligned(32))) _Float16 sA[8 * 8 * 512]; // 64 KB
    const int tid  = threadIdx.x;
    const int lane = tid & 31;
    const int w    = tid >> 5;
    const int hi   = lane >> 4;
    const int n    = lane & 15;

    // B fragments of w_down (K=c contiguous in memory): held in regs all kernel.
    v16h Bf[8];
#pragma unroll
    for (int ks = 0; ks < 8; ++ks) {
        const float* src = wdn + n * C + ks * 32 + hi * 8;
        v16h b;
#pragma unroll
        for (int h = 0; h < 16; ++h) b[h] = (_Float16)src[h + (h >> 3) * 8];
        Bf[ks] = b;
    }

    // Loop-invariant staging setup: this thread always covers channels c4..c4+3,
    // which land in the same lane/kk with consecutive h (8B-aligned LDS chunk).
    const int c4 = (tid & 63) << 2;
    int kkq, lnq, hq;
    afrag_pos(c4, 0, kkq, lnq, hq);
    const int hiq = lnq >> 4;
    const float m0 = stats[c4], m1 = stats[c4 + 1], m2 = stats[c4 + 2], m3 = stats[c4 + 3];
    const float i0 = stats[C + c4], i1 = stats[C + c4 + 1];
    const float i2 = stats[C + c4 + 2], i3 = stats[C + c4 + 3];

    for (int ci = 0; ci < 4; ++ci) {
        const int pbase = (blockIdx.x * 4 + ci) * 128;
        __syncthreads();
        // Stage 128 px x 256 ch: BN + relu + sigmoid -> packed f16 A-fragments.
#pragma unroll 4
        for (int i = 0; i < 32; ++i) {
            const int p = (tid >> 6) + (i << 2);
            const float* gp = x + (size_t)(pbase + p) * C + c4;
            if (ci < 3) __builtin_prefetch(gp + 128 * C, 0, 1); // global_prefetch_b8
            float4 v = *(const float4*)gp;
            float x0 = (v.x - m0) * i0, x1 = (v.y - m1) * i1;
            float x2 = (v.z - m2) * i2, x3 = (v.w - m3) * i3;
            union { _Float16 hh[4]; uint2 uu; } pk;
            pk.hh[0] = (_Float16)(x0 > 0.f ? 1.f / (1.f + __expf(-x0)) : 0.5f);
            pk.hh[1] = (_Float16)(x1 > 0.f ? 1.f / (1.f + __expf(-x1)) : 0.5f);
            pk.hh[2] = (_Float16)(x2 > 0.f ? 1.f / (1.f + __expf(-x2)) : 0.5f);
            pk.hh[3] = (_Float16)(x3 > 0.f ? 1.f / (1.f + __expf(-x3)) : 0.5f);
            const int idx = ((((p >> 4) * 8 + kkq) * 32) + (p & 15) + 16 * hiq) * 16 + hq;
            *(uint2*)(sA + idx) = pk.uu;   // ds_store_b64
        }
        __syncthreads();
        // Wave w computes its 16-pixel tile: K=256 in 8 WMMA steps.
        v8f acc = {0.f, 0.f, 0.f, 0.f, 0.f, 0.f, 0.f, 0.f};
        const v16h* Af = (const v16h*)(sA + (size_t)w * 8 * 512);
#pragma unroll
        for (int ks = 0; ks < 8; ++ks)
            acc = wmma_f16(Af[ks * 32 + lane], Bf[ks], acc);
        // Store y: one base, constant 64B offsets.
        float* dst = y + (size_t)(pbase + w * 16 + 8 * hi) * 16 + n;
#pragma unroll
        for (int j = 0; j < 8; ++j) dst[j * 16] = acc[j];
    }
}

// K4: Gram of sample 0: G[i,j] = sum_{p<HW} y[p,i]*y[p,j]  (tiny, 16x16).
__global__ void gram_k(const float* __restrict__ y, float* __restrict__ G) {
    const int t = threadIdx.x, i = t >> 4, j = t & 15;
    float s = 0.f;
    for (int p = blockIdx.x; p < HW; p += gridDim.x)
        s += y[(size_t)p * 16 + i] * y[(size_t)p * 16 + j];
    atomicAdd(&G[i * 16 + j], s);
}

// K5: M2 = Wf2 @ (w_up @ G0); pack M2 and Wf1 into B-fragment layouts in ws.
__global__ void prep_k(const float* __restrict__ w_up,
                       const float* __restrict__ w_final,
                       const float* __restrict__ G,
                       _Float16* __restrict__ m2f,
                       _Float16* __restrict__ wf1f) {
    __shared__ float sT[256][16];
    const int t = threadIdx.x;
    // T = w_up @ G0   [256 x 16]
    {
        float acc[16] = {};
#pragma unroll
        for (int o = 0; o < 16; ++o) {
            float wu = w_up[t * 16 + o];
#pragma unroll
            for (int j = 0; j < 16; ++j) acc[j] += wu * G[o * 16 + j];
        }
#pragma unroll
        for (int j = 0; j < 16; ++j) sT[t][j] = acc[j];
    }
    __syncthreads();
    // M2[t,:] = Wf2[t,:] @ T  ->  pack into B-fragment layout (K padded to 32).
    {
        float acc[16] = {};
        const float* wf2 = w_final + (size_t)t * 512 + 256;
        for (int c = 0; c < 256; ++c) {
            float wv = wf2[c];
#pragma unroll
            for (int j = 0; j < 16; ++j) acc[j] += wv * sT[c][j];
        }
        const int nt = t >> 4, nn = t & 15;
#pragma unroll
        for (int kr = 0; kr < 32; ++kr) {
            int hi = (kr >> 3) & 1, h = (kr & 7) + ((kr >> 4) << 3);
            int ln = nn + 16 * hi;
            m2f[((nt * 32) + ln) * 16 + h] =
                (kr < 16) ? (_Float16)acc[kr] : (_Float16)0.f;
        }
    }
    // Wf1 (= w_final[:, :256]) B-fragment pack: thread t owns output row o=t.
    {
        const float* wrow = w_final + (size_t)t * 512;
        const int nt = t >> 4, nn = t & 15;
        for (int k = 0; k < 256; ++k) {
            int kk = k >> 5, kr = k & 31, hi = (kr >> 3) & 1, h = (kr & 7) + ((kr >> 4) << 3);
            int ln = nn + 16 * hi;
            wf1f[(((kk * 16 + nt) * 32) + ln) * 16 + h] = (_Float16)wrow[k];
        }
    }
}

// K6: fused final GEMM: out[p,:] = Wf1 @ xn[p,:] + M2 @ y[p,:]  (NCHW output).
// 256 threads = 8 waves. 144KB of f16 weight fragments are DMA'd into LDS by
// the Tensor Data Mover once per workgroup; each 64-pixel chunk stages f16 xn
// A-fragments; 9 WMMAs per 16x16 output tile.
__global__ void final_gemm_k(const float* __restrict__ x,
                             const float* __restrict__ stats,
                             const float* __restrict__ yws,
                             const _Float16* __restrict__ wf1f,  // +m2f contiguous
                             float* __restrict__ out) {
    __shared__ __attribute__((aligned(32))) _Float16 sWM[73728]; // 144 KB frags
    __shared__ __attribute__((aligned(32))) _Float16 sA[16384];  //  32 KB xn frags
    __shared__ float sT[8][16][17];                              // per-wave transpose
    const int tid  = threadIdx.x;
    const int lane = tid & 31;
    const int w    = tid >> 5;
    const int hi   = lane >> 4;
    const int n    = lane & 15;

    // --- TDM: one 2D descriptor copies wf1f+m2f (4096 dwords x 9 rows) to LDS.
    if (w == 0) {
        const uint32_t lds = (uint32_t)(uintptr_t)sWM;  // low 32b of LDS aperture ptr
        const uint64_t ga  = (uint64_t)(uintptr_t)wf1f;
        u32x4 g0;
        g0[0] = 1u;                                     // count=1 (valid descriptor)
        g0[1] = lds;                                    // lds_addr
        g0[2] = (uint32_t)ga;                           // global_addr[31:0]
        g0[3] = (uint32_t)(ga >> 32) | (2u << 30);      // global_addr[56:32] | type=2
        u32x8 g1;
        g1[0] = 2u << 16;                               // data_size = 4 bytes
        g1[1] = 4096u << 16;                            // tensor_dim0 = 4096 dwords
        g1[2] = 9u << 16;                               // tensor_dim1 = 9
        g1[3] = 4096u << 16;                            // tile_dim0 = 4096
        g1[4] = 9u;                                     // tile_dim1 = 9
        g1[5] = 4096u;                                  // tensor_dim0_stride = 4096
        g1[6] = 0u;
        g1[7] = 0u;
        asm volatile("tensor_load_to_lds %0, %1" :: "s"(g0), "s"(g1));
        __builtin_amdgcn_s_wait_tensorcnt(0);
    }

    // Loop-invariant staging setup (same hoisting as down_gemm_k).
    const int c4 = (tid & 63) << 2;
    int kkq, lnq, hq;
    afrag_pos(c4, 0, kkq, lnq, hq);
    const int hiq = lnq >> 4;
    const float m0 = stats[c4], m1 = stats[c4 + 1], m2 = stats[c4 + 2], m3 = stats[c4 + 3];
    const float i0 = stats[C + c4], i1 = stats[C + c4 + 1];
    const float i2 = stats[C + c4 + 2], i3 = stats[C + c4 + 3];

    for (int ci = 0; ci < 8; ++ci) {
        const size_t P0 = ((size_t)blockIdx.x * 8 + ci) * 64;
        __syncthreads();   // also publishes the TDM-filled weights on ci==0
        // Stage 64 px of xn as packed f16 A-fragments.
#pragma unroll 4
        for (int i = 0; i < 16; ++i) {
            const int p = (tid >> 6) + (i << 2);
            const float* gp = x + (P0 + p) * C + c4;
            if (ci < 7) __builtin_prefetch(gp + 64 * C, 0, 1);
            float4 v = *(const float4*)gp;
            union { _Float16 hh[4]; uint2 uu; } pk;
            pk.hh[0] = (_Float16)((v.x - m0) * i0);
            pk.hh[1] = (_Float16)((v.y - m1) * i1);
            pk.hh[2] = (_Float16)((v.z - m2) * i2);
            pk.hh[3] = (_Float16)((v.w - m3) * i3);
            const int idx = ((((p >> 4) * 8 + kkq) * 32) + (p & 15) + 16 * hiq) * 16 + hq;
            *(uint2*)(sA + idx) = pk.uu;   // ds_store_b64
        }
        __syncthreads();

        const int pt = w & 3;          // pixel tile within chunk
        const int oh = w >> 2;         // output half (8 tiles each)

        // A fragments (full K=256) held in registers, reused across 8 N-tiles.
        v16h A[8];
        const v16h* Af = (const v16h*)(sA + (size_t)pt * 8 * 512);
#pragma unroll
        for (int kk = 0; kk < 8; ++kk) A[kk] = Af[kk * 32 + lane];

        // y A-fragment (K=16 padded to 32): lane loads 8 contiguous floats.
        v16h Y;
        {
            const size_t yb = (P0 + pt * 16 + n) * 16 + hi * 8;
            float4 y0 = *(const float4*)(yws + yb);
            float4 y1 = *(const float4*)(yws + yb + 4);
            Y[0] = (_Float16)y0.x; Y[1] = (_Float16)y0.y;
            Y[2] = (_Float16)y0.z; Y[3] = (_Float16)y0.w;
            Y[4] = (_Float16)y1.x; Y[5] = (_Float16)y1.y;
            Y[6] = (_Float16)y1.z; Y[7] = (_Float16)y1.w;
#pragma unroll
            for (int h = 8; h < 16; ++h) Y[h] = (_Float16)0.f;
        }

        const int b   = (int)(P0 >> 16);                // sample
        const int pin = (int)(P0 & (HW - 1)) + pt * 16; // pixel within sample
        const v16h* Wv = (const v16h*)sWM;
        const v16h* Mv = (const v16h*)(sWM + 65536);

        for (int ot8 = 0; ot8 < 8; ++ot8) {
            const int ot = oh * 8 + ot8;
            v8f acc = {0.f, 0.f, 0.f, 0.f, 0.f, 0.f, 0.f, 0.f};
#pragma unroll
            for (int kk = 0; kk < 8; ++kk)
                acc = wmma_f16(A[kk], Wv[((kk * 16 + ot) * 32) + lane], acc);
            acc = wmma_f16(Y, Mv[ot * 32 + lane], acc);   // rank-16 M2 term

            // Transpose 16x16 tile through LDS -> coalesced NCHW b128 stores.
#pragma unroll
            for (int j = 0; j < 8; ++j) sT[w][j + 8 * hi][n] = acc[j];
            __syncthreads();
            {
                const int ol  = lane >> 1;
                const int pxb = (lane & 1) * 8;
                float4 o0 = {sT[w][pxb + 0][ol], sT[w][pxb + 1][ol],
                             sT[w][pxb + 2][ol], sT[w][pxb + 3][ol]};
                float4 o1 = {sT[w][pxb + 4][ol], sT[w][pxb + 5][ol],
                             sT[w][pxb + 6][ol], sT[w][pxb + 7][ol]};
                float* dst = out + ((size_t)(b * C + ot * 16 + ol)) * HW + pin + pxb;
                *(float4*)dst       = o0;   // global_store_b128
                *(float4*)(dst + 4) = o1;
            }
            __syncthreads();
        }
    }
}

// ---------------------------------------------------------------------------
extern "C" void kernel_launch(void* const* d_in, const int* in_sizes, int n_in,
                              void* d_out, int out_size, void* d_ws, size_t ws_size,
                              hipStream_t stream) {
    const float* x       = (const float*)d_in[0];  // [2,256,256,256] NHWC
    const float* w_down  = (const float*)d_in[1];  // [16,256]
    const float* w_up    = (const float*)d_in[2];  // [256,16]
    const float* w_final = (const float*)d_in[3];  // [256,512]
    float* out = (float*)d_out;                    // [2,256,256,256] NCHW

    float*     stats = (float*)((char*)d_ws + WS_STATS);
    float*     G     = (float*)((char*)d_ws + WS_G);
    _Float16*  wf1f  = (_Float16*)((char*)d_ws + WS_WF1F);
    _Float16*  m2f   = (_Float16*)((char*)d_ws + WS_M2F);
    float*     yws   = (float*)((char*)d_ws + WS_Y);

    init_k       <<<1,   256, 0, stream>>>(stats);
    bn_stats_k   <<<512, 256, 0, stream>>>(x, stats);
    bn_finalize_k<<<1,   256, 0, stream>>>(stats);
    down_gemm_k  <<<256, 256, 0, stream>>>(x, w_down, stats, yws);
    gram_k       <<<64,  256, 0, stream>>>(yws, G);
    prep_k       <<<1,   256, 0, stream>>>(w_up, w_final, G, m2f, wf1f);
    final_gemm_k <<<256, 256, 0, stream>>>(x, stats, yws, wf1f, out);
}
